// MultiModalFusion_30253749633498
// MI455X (gfx1250) — matrix operand order repaired and checked by reading, not compiled, not run on hardware
//
#include <hip/hip_runtime.h>
#include <hip/hip_bf16.h>

// Problem constants
#define B_ 4096
#define L_ 8
#define D_ 256
#define O_ 256
#define E_ 64
#define C_ 16
#define H_ 8
// head dim = 32

typedef _Float16 h16;
typedef _Float16 v16h __attribute__((ext_vector_type(16)));
typedef _Float16 v8h  __attribute__((ext_vector_type(8)));
typedef float    v8f  __attribute__((ext_vector_type(8)));

// ---------------------------------------------------------------------------
// fp32 -> f16 convert
// ---------------------------------------------------------------------------
__global__ void k_f32_to_f16(const float* __restrict__ in, h16* __restrict__ out, int n) {
    int i = blockIdx.x * 256 + threadIdx.x;
    if (i < n) out[i] = (h16)in[i];
}

// Wa [C,D,O] fp32 -> WaT [C,O,D] f16   (so WMMA B-frag reads contiguous K)
__global__ void k_wa_transpose(const float* __restrict__ Wa, h16* __restrict__ WaT) {
    int i = blockIdx.x * 256 + threadIdx.x;        // 16*256*256 = 1048576
    int c = i >> 16;
    int r = i & 65535;
    int n = r >> 8;
    int k = r & 255;
    WaT[i] = (h16)Wa[(c << 16) + (k << 8) + n];
}

// e_cls[c][l][m] = sum_e edge_weights[c,l,m,e] * a_edge[c,e]
__global__ void k_ecls(const float* __restrict__ ew, const float* __restrict__ ae,
                       float* __restrict__ ecls) {
    int i = blockIdx.x * 256 + threadIdx.x;        // C*L*L = 1024
    if (i >= C_ * L_ * L_) return;
    int c = i >> 6;
    const float* p = ew + (size_t)i * E_;
    const float* a = ae + c * E_;
    float s = 0.f;
#pragma unroll
    for (int e = 0; e < E_; ++e) s += p[e] * a[e];
    ecls[i] = s;
}

// Stable counting sort positions: pos[b] = offset[label[b]] + stable rank
__global__ __launch_bounds__(256) void k_pos(const int* __restrict__ labels,
                                             int* __restrict__ pos) {
    __shared__ int cnt[256][C_];
    __shared__ int tot[C_];
    __shared__ int offs[C_];
    int t = threadIdx.x;
#pragma unroll
    for (int c = 0; c < C_; ++c) cnt[t][c] = 0;
    for (int i = 0; i < 16; ++i) cnt[t][labels[t * 16 + i]]++;
    __syncthreads();
    if (t < C_) {                                   // exclusive scan per class
        int run = 0;
        for (int j = 0; j < 256; ++j) { int v = cnt[j][t]; cnt[j][t] = run; run += v; }
        tot[t] = run;
    }
    __syncthreads();
    if (t == 0) {
        int run = 0;
        for (int c = 0; c < C_; ++c) { offs[c] = run; run += tot[c]; }
    }
    __syncthreads();
    for (int i = 0; i < 16; ++i) {
        int b = t * 16 + i;
        int c = labels[b];
        pos[b] = offs[c] + cnt[t][c];
        cnt[t][c]++;
    }
}

// ---------------------------------------------------------------------------
// WMMA GEMM:  out[M=32768, N=256] = X[M,256] @ W^T + bias, out in f16
// W is row-major [N, K] in f16. AF32: X is fp32 (converted in the A-frag load).
// grid = (512, 4), block = 128 (4 waves).
// Each wave computes a 64x16 output strip: 4 M-tiles sharing one B fragment
// per k-step -> 4 independent WMMA chains, 4x B reuse.
// ---------------------------------------------------------------------------
template <bool AF32>
__global__ __launch_bounds__(128) void k_gemm16(const void* __restrict__ Xv,
                                                const h16* __restrict__ W,
                                                const float* __restrict__ bias,
                                                h16* __restrict__ out) {
    int tid = threadIdx.x;
    int wv = tid >> 5, lane = tid & 31;
    int lm = lane & 15, hi = (lane >> 4) & 1;
    int m_base = blockIdx.x << 6;                       // 64 rows per block/wave
    int n0 = (blockIdx.y << 6) + (wv << 4);

    v8f acc[4];
#pragma unroll
    for (int mt = 0; mt < 4; ++mt)
#pragma unroll
        for (int j = 0; j < 8; ++j) acc[mt][j] = 0.f;

    const h16* wrow = W + (size_t)(n0 + lm) * D_ + hi * 8;    // B: table-order K runs

#pragma unroll 2
    for (int k0 = 0; k0 < D_; k0 += 32) {
        v8h b0 = *(const v8h*)(wrow + k0);
        v8h b1 = *(const v8h*)(wrow + k0 + 16);
        v16h bb = __builtin_shufflevector(b0, b1, 0, 1, 2, 3, 4, 5, 6, 7,
                                          8, 9, 10, 11, 12, 13, 14, 15);
#pragma unroll
        for (int mt = 0; mt < 4; ++mt) {
            int row = m_base + (mt << 4) + lm;
            v16h a;
            if (AF32) {
                const float4* x4 =
                    (const float4*)((const float*)Xv + (size_t)row * D_ + k0 + hi * 8);
                float4 f0 = x4[0], f1 = x4[1], f2 = x4[4], f3 = x4[5];
                a[0] = (h16)f0.x;  a[1] = (h16)f0.y;  a[2] = (h16)f0.z;  a[3] = (h16)f0.w;
                a[4] = (h16)f1.x;  a[5] = (h16)f1.y;  a[6] = (h16)f1.z;  a[7] = (h16)f1.w;
                a[8] = (h16)f2.x;  a[9] = (h16)f2.y;  a[10] = (h16)f2.z; a[11] = (h16)f2.w;
                a[12] = (h16)f3.x; a[13] = (h16)f3.y; a[14] = (h16)f3.z; a[15] = (h16)f3.w;
            } else {
                const h16* xr = (const h16*)Xv + (size_t)row * D_ + k0 + hi * 8;
                v8h lo = *(const v8h*)xr;
                v8h hh = *(const v8h*)(xr + 16);
                a = __builtin_shufflevector(lo, hh, 0, 1, 2, 3, 4, 5, 6, 7,
                                            8, 9, 10, 11, 12, 13, 14, 15);
            }
            acc[mt] = __builtin_amdgcn_wmma_f32_16x16x32_f16(false, a, false, bb,
                                                             (short)0, acc[mt],
                                                             false, false);
        }
    }

    int n = n0 + lm;
    float bn = bias[n];
#pragma unroll
    for (int mt = 0; mt < 4; ++mt) {
        int rbase = m_base + (mt << 4) + hi * 8;
#pragma unroll
        for (int j = 0; j < 8; ++j)
            out[(size_t)(rbase + j) * D_ + n] = (h16)(acc[mt][j] + bn);
    }
}

// ---------------------------------------------------------------------------
// Per-(b,head) attention: 8x8 scores, softmax, ctx = attn @ v  (f16 out)
// grid = B, block = 256 (8 waves, wave == head)
// ---------------------------------------------------------------------------
__global__ __launch_bounds__(256) void k_attn(const h16* __restrict__ Q,
                                              const h16* __restrict__ K,
                                              const h16* __restrict__ V,
                                              h16* __restrict__ ctx) {
    int b = blockIdx.x, tid = threadIdx.x;
    int w = tid >> 5, lane = tid & 31;
    __shared__ float sc[H_][L_][L_];
    size_t base = (size_t)b * L_ * D_ + w * 32;
#pragma unroll
    for (int rep = 0; rep < 2; ++rep) {
        int p = lane + rep * 32;
        int l = p >> 3, m = p & 7;
        const v8h* qr = (const v8h*)(Q + base + (size_t)l * D_);
        const v8h* kr = (const v8h*)(K + base + (size_t)m * D_);
        float s = 0.f;
#pragma unroll
        for (int j = 0; j < 4; ++j) {
            v8h qv = qr[j], kv = kr[j];
#pragma unroll
            for (int i = 0; i < 8; ++i) s += (float)qv[i] * (float)kv[i];
        }
        sc[w][l][m] = s * 0.17677669529663687f;    // 1/sqrt(32)
    }
    __syncthreads();
    if (lane < 8) {
        int l = lane;
        float s[8], mx = -1e30f;
#pragma unroll
        for (int m = 0; m < 8; ++m) { s[m] = sc[w][l][m]; mx = fmaxf(mx, s[m]); }
        float sum = 0.f;
#pragma unroll
        for (int m = 0; m < 8; ++m) { s[m] = expf(s[m] - mx); sum += s[m]; }
        float inv = 1.f / sum;
#pragma unroll
        for (int m = 0; m < 8; ++m) sc[w][l][m] = s[m] * inv;
    }
    __syncthreads();
    {
        int d = lane;
        float vv[8];
#pragma unroll
        for (int m = 0; m < 8; ++m) vv[m] = (float)V[base + (size_t)m * D_ + d];
#pragma unroll
        for (int l = 0; l < 8; ++l) {
            float a = 0.f;
#pragma unroll
            for (int m = 0; m < 8; ++m) a += sc[w][l][m] * vv[m];
            ctx[base + (size_t)l * D_ + d] = (h16)a;
        }
    }
}

// ---------------------------------------------------------------------------
// Per-b GAT expert (only the class this row actually needs).
// grid = B, block = 256 (8 waves; each wave owns a 32-column strip of Wh).
// A_F rows are staged once into LDS (rows 8..15 zeroed); the k-loop issues
// both N-tiles' WMMAs per k-step on independent accumulator chains, sharing
// one A fragment (conflict-free ds_load_b128 pair) per step.
// ---------------------------------------------------------------------------
#define AP_ 264   // padded LDS row stride in halves (528B: rows hit distinct banks)

__global__ __launch_bounds__(256) void k_gat(const h16* __restrict__ AF,
                                             const h16* __restrict__ WaT,
                                             const float* __restrict__ a_src,
                                             const float* __restrict__ a_dst,
                                             const float* __restrict__ ecls,
                                             const int* __restrict__ labels,
                                             const int* __restrict__ pos,
                                             float* __restrict__ out) {
    int b = blockIdx.x;
    int c = labels[b];
    int tid = threadIdx.x;
    int wv = tid >> 5, lane = tid & 31;
    int lm = lane & 15, hi = (lane >> 4) & 1;

    __shared__ h16  Atile[16 * AP_];   // 8.25 KB: A_F[b] (rows 0..7) + zero pad rows
    __shared__ float Wh[L_][O_];       // 8 KB
    __shared__ float att[L_][L_];
    __shared__ float srcv[L_], dstv[L_];

    // ---- stage A_F[b] (8 x 256 f16) into LDS; zero rows 8..15 ----
    {
        int r = tid >> 5;                 // 0..7
        int col = (tid & 31) * 8;         // 0..248
        *(v8h*)&Atile[r * AP_ + col] =
            *(const v8h*)(AF + (size_t)(b * L_ + r) * D_ + col);
        v8h z;
#pragma unroll
        for (int i = 0; i < 8; ++i) z[i] = (h16)0.f;
        *(v8h*)&Atile[(r + 8) * AP_ + col] = z;
    }
    __syncthreads();

    const h16* WT = WaT + ((size_t)c << 16);       // WaT[c] : [O, D] f16
    int n0 = wv << 5;                              // 32-column strip per wave
    const h16* wr0 = WT + (size_t)(n0 + lm) * D_ + hi * 8;
    const h16* wr1 = wr0 + (size_t)16 * D_;
    const h16* ar = &Atile[lm * AP_ + hi * 8];

    v8f acc0, acc1;
#pragma unroll
    for (int j = 0; j < 8; ++j) { acc0[j] = 0.f; acc1[j] = 0.f; }

#pragma unroll
    for (int k0 = 0; k0 < D_; k0 += 32) {
        v8h lo = *(const v8h*)(ar + k0);
        v8h hh = *(const v8h*)(ar + k0 + 16);
        v16h a = __builtin_shufflevector(lo, hh, 0, 1, 2, 3, 4, 5, 6, 7,
                                         8, 9, 10, 11, 12, 13, 14, 15);
        v8h b00 = *(const v8h*)(wr0 + k0);
        v8h b01 = *(const v8h*)(wr0 + k0 + 16);
        v8h b10 = *(const v8h*)(wr1 + k0);
        v8h b11 = *(const v8h*)(wr1 + k0 + 16);
        v16h bb0 = __builtin_shufflevector(b00, b01, 0, 1, 2, 3, 4, 5, 6, 7,
                                           8, 9, 10, 11, 12, 13, 14, 15);
        v16h bb1 = __builtin_shufflevector(b10, b11, 0, 1, 2, 3, 4, 5, 6, 7,
                                           8, 9, 10, 11, 12, 13, 14, 15);
        acc0 = __builtin_amdgcn_wmma_f32_16x16x32_f16(false, a, false, bb0,
                                                      (short)0, acc0, false, false);
        acc1 = __builtin_amdgcn_wmma_f32_16x16x32_f16(false, a, false, bb1,
                                                      (short)0, acc1, false, false);
    }
    if (!hi) {                                    // lanes 0..15 hold rows 0..7
#pragma unroll
        for (int j = 0; j < 8; ++j) {
            Wh[j][n0 + lm]      = acc0[j];
            Wh[j][n0 + 16 + lm] = acc1[j];
        }
    }
    __syncthreads();

    // src/dst: wave w reduces row l = w  (deterministic shuffle tree)
    {
        int l = wv;
        const float* as = a_src + c * O_;
        const float* ad = a_dst + c * O_;
        float ps = 0.f, pd = 0.f;
#pragma unroll
        for (int i = 0; i < 8; ++i) {
            int n = lane * 8 + i;
            float whv = Wh[l][n];
            ps += whv * as[n];
            pd += whv * ad[n];
        }
#pragma unroll
        for (int off = 16; off; off >>= 1) {
            ps += __shfl_xor(ps, off);
            pd += __shfl_xor(pd, off);
        }
        if (lane == 0) { srcv[l] = ps; dstv[l] = pd; }
    }
    __syncthreads();

    if (tid < 8) {                                   // tanh + softmax over m
        int l = tid;
        float s[8], mx = -1e30f;
#pragma unroll
        for (int m = 0; m < 8; ++m) {
            s[m] = tanhf(srcv[l] + dstv[m] + ecls[c * 64 + l * 8 + m]);
            mx = fmaxf(mx, s[m]);
        }
        float sum = 0.f;
#pragma unroll
        for (int m = 0; m < 8; ++m) { s[m] = expf(s[m] - mx); sum += s[m]; }
        float inv = 1.f / sum;
#pragma unroll
        for (int m = 0; m < 8; ++m) att[l][m] = s[m] * inv;
    }
    __syncthreads();

    {                                                // mean_l tanh(att @ Wh)
        int n = tid;
        float acc = 0.f;
#pragma unroll
        for (int l = 0; l < 8; ++l) {
            float t = 0.f;
#pragma unroll
            for (int m = 0; m < 8; ++m) t += att[l][m] * Wh[m][n];
            acc += tanhf(t);
        }
        out[(size_t)pos[b] * O_ + n] = acc * 0.125f; // scatter into sorted slot
    }
}

// ---------------------------------------------------------------------------
extern "C" void kernel_launch(void* const* d_in, const int* in_sizes, int n_in,
                              void* d_out, int out_size, void* d_ws, size_t ws_size,
                              hipStream_t stream) {
    const float* A_I    = (const float*)d_in[0];
    const float* A_V    = (const float*)d_in[1];
    const int*   labels = (const int*)d_in[2];
    const float* ipw    = (const float*)d_in[3];
    const float* ipb    = (const float*)d_in[4];
    const float* opw    = (const float*)d_in[5];
    const float* opb    = (const float*)d_in[6];
    const float* Wa     = (const float*)d_in[7];
    const float* a_src  = (const float*)d_in[8];
    const float* a_dst  = (const float*)d_in[9];
    const float* a_edge = (const float*)d_in[10];
    const float* ew     = (const float*)d_in[11];
    float* out = (float*)d_out;

    const size_t BLD = (size_t)B_ * L_ * D_;        // 8,388,608

    char* p = (char*)d_ws;
    auto carve = [&](size_t bytes) {
        char* r = p;
        p += (bytes + 255) & ~(size_t)255;
        return r;
    };
    h16*   Q16   = (h16*)carve(BLD * 2);
    h16*   K16   = (h16*)carve(BLD * 2);
    h16*   V16   = (h16*)carve(BLD * 2);
    h16*   ctx16 = (h16*)carve(BLD * 2);
    h16*   AF16  = (h16*)carve(BLD * 2);
    h16*   ipw16 = (h16*)carve((size_t)3 * D_ * D_ * 2);
    h16*   opw16 = (h16*)carve((size_t)D_ * D_ * 2);
    h16*   WaT16 = (h16*)carve((size_t)C_ * D_ * O_ * 2);
    float* ecls  = (float*)carve((size_t)C_ * L_ * L_ * 4);
    int*   pos   = (int*)carve((size_t)B_ * 4);

    // Weight prep
    k_f32_to_f16<<<dim3(768), dim3(256), 0, stream>>>(ipw, ipw16, 3 * D_ * D_);
    k_f32_to_f16<<<dim3(256), dim3(256), 0, stream>>>(opw, opw16, D_ * D_);
    k_wa_transpose<<<dim3(4096), dim3(256), 0, stream>>>(Wa, WaT16);
    k_ecls<<<dim3(4), dim3(256), 0, stream>>>(ew, a_edge, ecls);
    k_pos<<<dim3(1), dim3(256), 0, stream>>>(labels, pos);

    // QKV projections (fp32 A, converted in the A-fragment load)
    dim3 ggrid(B_ * L_ / 64, D_ / 64);               // (512, 4)
    k_gemm16<true><<<ggrid, dim3(128), 0, stream>>>(A_I, ipw16,             ipb,        Q16);
    k_gemm16<true><<<ggrid, dim3(128), 0, stream>>>(A_V, ipw16 + D_ * D_,   ipb + D_,   K16);
    k_gemm16<true><<<ggrid, dim3(128), 0, stream>>>(A_V, ipw16 + 2*D_*D_,   ipb + 2*D_, V16);

    // Attention (per batch row, 8 heads)
    k_attn<<<dim3(B_), dim3(256), 0, stream>>>(Q16, K16, V16, ctx16);

    // Output projection (f16 A)
    k_gemm16<false><<<ggrid, dim3(128), 0, stream>>>(ctx16, opw16, opb, AF16);

    // Per-row GAT expert + scatter into stable-sorted order
    k_gat<<<dim3(B_), dim3(256), 0, stream>>>(AF16, WaT16, a_src, a_dst, ecls,
                                              labels, pos, out);
}
